// AdaptiveGeometryAttention_69398081569238
// MI455X (gfx1250) — matrix operand order repaired and checked by reading, not compiled, not run on hardware
//
#include <hip/hip_runtime.h>
#include <math.h>

#define B_ 2
#define T_ 1024
#define E_ 1024
#define H_ 16
#define D_ 64

typedef __attribute__((ext_vector_type(16))) __bf16 v16bf;
typedef __attribute__((ext_vector_type(8)))  __bf16 v8bf;
typedef __attribute__((ext_vector_type(8)))  float  v8f;

union V16U { v16bf v; v8bf h[2]; };

static __device__ __forceinline__ v8f wmma_bf(v16bf a, v16bf b, v8f c) {
  return __builtin_amdgcn_wmma_f32_16x16x32_bf16(false, a, false, b, (short)0, c, false, false);
}

// f32 -> bf16 (round-to-nearest-even), stored as ushort
static __device__ __forceinline__ unsigned short f2bfu(float f) {
  unsigned u = __float_as_uint(f);
  u += 0x7FFFu + ((u >> 16) & 1u);
  return (unsigned short)(u >> 16);
}
static __device__ __forceinline__ float bfu2f(unsigned short s) {
  return __uint_as_float(((unsigned)s) << 16);
}

// reductions across a 16-lane half of the wave (rows of the C fragment)
static __device__ __forceinline__ float hred_max(float x) {
  x = fmaxf(x, __shfl_xor(x, 1, 32));
  x = fmaxf(x, __shfl_xor(x, 2, 32));
  x = fmaxf(x, __shfl_xor(x, 4, 32));
  x = fmaxf(x, __shfl_xor(x, 8, 32));
  return x;
}
static __device__ __forceinline__ float hred_sum(float x) {
  x += __shfl_xor(x, 1, 32);
  x += __shfl_xor(x, 2, 32);
  x += __shfl_xor(x, 4, 32);
  x += __shfl_xor(x, 8, 32);
  return x;
}

// ---------------------------------------------------------------- convert
__global__ void k_f32_to_bf16(const float* __restrict__ src,
                              unsigned short* __restrict__ dst, int n) {
  int i = blockIdx.x * blockDim.x + threadIdx.x;
  if (i < n) dst[i] = f2bfu(src[i]);
}

// ---------------------------------------------------- importance + alpha
__global__ void k_imp_alpha(const float* __restrict__ x,
                            const float* __restrict__ Wimp,
                            const float* __restrict__ bimp,
                            const float* __restrict__ Walpha,
                            const float* __restrict__ balpha,
                            const float* __restrict__ thr,
                            float* __restrict__ alpha,
                            float* __restrict__ spike) {
  int row = blockIdx.x;  // b*T + t
  const float* xr = x + (size_t)row * E_;
  __shared__ float red[17][256];
  float acc[17];
#pragma unroll
  for (int o = 0; o < 17; o++) acc[o] = 0.f;
  for (int i = threadIdx.x; i < E_; i += 256) {
    float xv = xr[i];
    acc[0] += xv * Wimp[i];
#pragma unroll
    for (int hh = 0; hh < 16; hh++) acc[1 + hh] += xv * Walpha[hh * E_ + i];
  }
#pragma unroll
  for (int o = 0; o < 17; o++) red[o][threadIdx.x] = acc[o];
  __syncthreads();
  if (threadIdx.x < 17) {
    float s = 0.f;
    for (int i = 0; i < 256; i++) s += red[threadIdx.x][i];
    if (threadIdx.x == 0) {
      float imp = 1.f / (1.f + expf(-(s + bimp[0])));
      spike[row] = (imp > thr[0]) ? 1.f : 0.f;
    } else {
      int hh = threadIdx.x - 1;
      alpha[(size_t)row * H_ + hh] = 1.f / (1.f + expf(-(s + balpha[hh])));
    }
  }
}

// ------------------------------------------------------------- QKV GEMM
// out[row, n] = x[row,:] . Wqkv[n,:] + bqkv[n];  scatter to q/k [B,H,T,D]
// and v transposed [B,H,D,T].
__global__ void __launch_bounds__(256)
k_qkv_gemm(const unsigned short* __restrict__ xbf,
           const unsigned short* __restrict__ wbf,
           const float* __restrict__ bqkv,
           unsigned short* __restrict__ qbf,
           unsigned short* __restrict__ kbf,
           unsigned short* __restrict__ vt) {
  int tile = blockIdx.x * 8 + (threadIdx.x >> 5);   // 6144 tiles
  int lane = threadIdx.x & 31;
  const int NT = (3 * E_) / 64;                     // 48
  int mtile = tile / NT;
  int nt64  = tile % NT;
  int rowb = mtile * 16, colb = nt64 * 64;
  int grp = lane >> 4, ln = lane & 15;
  int ka = grp ? 16 : 0, kb16 = grp ? 16 : 0;

  v8f c0 = {}, c1 = {}, c2 = {}, c3 = {};
  const unsigned short* arow = xbf + (size_t)(rowb + ln) * E_;
  for (int kk = 0; kk < E_; kk += 32) {
    __builtin_prefetch(arow + kk + 128, 0, 1);
    V16U a;
    a.h[0] = *(const v8bf*)(arow + kk + ka);
    a.h[1] = *(const v8bf*)(arow + kk + 8 + ka);
    const unsigned short* bb = wbf + (size_t)colb * E_ + kk + kb16;
    v16bf b0 = *(const v16bf*)(bb + (size_t)(0 * 16 + ln) * E_);
    v16bf b1 = *(const v16bf*)(bb + (size_t)(1 * 16 + ln) * E_);
    v16bf b2 = *(const v16bf*)(bb + (size_t)(2 * 16 + ln) * E_);
    v16bf b3 = *(const v16bf*)(bb + (size_t)(3 * 16 + ln) * E_);
    c0 = wmma_bf(a.v, b0, c0);
    c1 = wmma_bf(a.v, b1, c1);
    c2 = wmma_bf(a.v, b2, c2);
    c3 = wmma_bf(a.v, b3, c3);
  }
#pragma unroll
  for (int nt = 0; nt < 4; nt++) {
    v8f c = (nt == 0) ? c0 : (nt == 1) ? c1 : (nt == 2) ? c2 : c3;
#pragma unroll
    for (int j = 0; j < 8; j++) {
      int row = rowb + grp * 8 + j;        // b*T + t
      int n = colb + nt * 16 + ln;         // 0..3071
      float val = c[j] + bqkv[n];
      int which = n >> 10;
      int cc = n & 1023;
      int hh = cc >> 6, d = cc & 63;
      int b = row >> 10, t = row & 1023;
      unsigned short bv = f2bfu(val);
      size_t bh = (size_t)(b * H_ + hh);
      if (which == 0)      qbf[(bh * T_ + t) * D_ + d] = bv;
      else if (which == 1) kbf[(bh * T_ + t) * D_ + d] = bv;
      else                 vt [(bh * D_ + d) * T_ + t] = bv;
    }
  }
}

// -------------------------------------------------------- hyperbolic map
__global__ void k_hyp(const unsigned short* __restrict__ src,
                      unsigned short* __restrict__ dst,
                      const float* __restrict__ qk_scale, int negfirst) {
  int row = blockIdx.x * blockDim.x + threadIdx.x;  // b*H*T rows
  if (row >= B_ * H_ * T_) return;
  const unsigned short* r = src + (size_t)row * D_;
  float u[D_];
  float nrm2 = 0.f;
#pragma unroll
  for (int i = 0; i < D_; i++) { u[i] = bfu2f(r[i]); nrm2 += u[i] * u[i]; }
  float ss = 1.5f / (1.f + expf(-qk_scale[0]));     // sigmoid(qk_scale)*1.5
  float inv = ss / fmaxf(sqrtf(nrm2), 1e-12f);
#pragma unroll
  for (int i = 0; i < D_; i++) u[i] *= inv;
  float mink = -u[0] * u[0];
#pragma unroll
  for (int i = 1; i < D_; i++) mink += u[i] * u[i];
  float nomin = sqrtf(fmaxf(mink, 1e-8f));
  float fac = sinhf(nomin) / nomin;
  float s2 = 0.f;
#pragma unroll
  for (int i = 1; i < D_; i++) { u[i] *= fac; s2 += u[i] * u[i]; }
  float tval = sqrtf(1.f + s2);
  unsigned short* o = dst + (size_t)row * D_;
  o[0] = f2bfu(negfirst ? -tval : tval);            // fold Minkowski sign
#pragma unroll
  for (int i = 1; i < D_; i++) o[i] = f2bfu(u[i]);
}

// ------------------------------------------------- flash-style attention
__global__ void __launch_bounds__(32)
k_attn(const unsigned short* __restrict__ qbf,
       const unsigned short* __restrict__ kbf,
       const unsigned short* __restrict__ qhbf,
       const unsigned short* __restrict__ khbf,
       const unsigned short* __restrict__ vt,
       const float* __restrict__ alpha,
       const float* __restrict__ spike,
       const float* __restrict__ log_k,
       unsigned short* __restrict__ ybf) {
  __shared__ unsigned short Pt[16 * 32];

  int gid = blockIdx.x;              // B*H*(T/16) = 2048
  int qt = gid & 63;
  int bh = gid >> 6;                 // 0..31
  int hh = bh & 15, b = bh >> 4;
  int lane = threadIdx.x & 31;
  int grp = lane >> 4, ln = lane & 15;
  int ka = grp ? 16 : 0, kc = grp ? 16 : 0;
  int qb = qt * 16;

  // A fragments for q (euclid) and q_hyp(-time) (hyper), K = 0..63
  const unsigned short* qrow  = qbf  + ((size_t)bh * T_ + qb + ln) * D_;
  const unsigned short* qhrow = qhbf + ((size_t)bh * T_ + qb + ln) * D_;
  V16U aq0, aq1, ah0, ah1;
  aq0.h[0] = *(const v8bf*)(qrow + 0 + ka);
  aq0.h[1] = *(const v8bf*)(qrow + 8 + ka);
  aq1.h[0] = *(const v8bf*)(qrow + 32 + ka);
  aq1.h[1] = *(const v8bf*)(qrow + 40 + ka);
  ah0.h[0] = *(const v8bf*)(qhrow + 0 + ka);
  ah0.h[1] = *(const v8bf*)(qhrow + 8 + ka);
  ah1.h[0] = *(const v8bf*)(qhrow + 32 + ka);
  ah1.h[1] = *(const v8bf*)(qhrow + 40 + ka);

  float curv = log1pf(expf(log_k[hh])) + 1e-6f;
  float aalpha[8];
#pragma unroll
  for (int j = 0; j < 8; j++) {
    int t = qb + grp * 8 + j;
    aalpha[j] = alpha[((size_t)b * T_ + t) * H_ + hh];
  }

  float m[8], l[8];
#pragma unroll
  for (int j = 0; j < 8; j++) { m[j] = -1e30f; l[j] = 0.f; }
  v8f acc0 = {}, acc1 = {}, acc2 = {}, acc3 = {};

  int niter = (qb + 47) >> 5;        // keys [0, qb+16) in 32-key chunks
  for (int it = 0; it < niter; it++) {
    int sb = it * 32;
    const unsigned short* kbase  = kbf  + ((size_t)bh * T_ + sb) * D_;
    const unsigned short* khbase = khbf + ((size_t)bh * T_ + sb) * D_;

    v8f se0 = {}, se1 = {}, sh0 = {}, sh1 = {};
    {
      v16bf b00 = *(const v16bf*)(kbase + (size_t)(ln) * D_ + 0 + kc);
      v16bf b01 = *(const v16bf*)(kbase + (size_t)(ln) * D_ + 32 + kc);
      v16bf b10 = *(const v16bf*)(kbase + (size_t)(16 + ln) * D_ + 0 + kc);
      v16bf b11 = *(const v16bf*)(kbase + (size_t)(16 + ln) * D_ + 32 + kc);
      se0 = wmma_bf(aq0.v, b00, se0); se0 = wmma_bf(aq1.v, b01, se0);
      se1 = wmma_bf(aq0.v, b10, se1); se1 = wmma_bf(aq1.v, b11, se1);
    }
    {
      v16bf b00 = *(const v16bf*)(khbase + (size_t)(ln) * D_ + 0 + kc);
      v16bf b01 = *(const v16bf*)(khbase + (size_t)(ln) * D_ + 32 + kc);
      v16bf b10 = *(const v16bf*)(khbase + (size_t)(16 + ln) * D_ + 0 + kc);
      v16bf b11 = *(const v16bf*)(khbase + (size_t)(16 + ln) * D_ + 32 + kc);
      sh0 = wmma_bf(ah0.v, b00, sh0); sh0 = wmma_bf(ah1.v, b01, sh0);
      sh1 = wmma_bf(ah0.v, b10, sh1); sh1 = wmma_bf(ah1.v, b11, sh1);
    }

#pragma unroll
    for (int j = 0; j < 8; j++) {
      int t = qb + grp * 8 + j;
      int s0 = sb + ln, s1 = sb + 16 + ln;
      float a = aalpha[j];
      float sc0, sc1;
      {
        float eu = se0[j] * 0.125f;                   // / sqrt(64)
        float mdot = fmaxf(-sh0[j], 1.000001f);
        float dd = acoshf(mdot);
        float hy = -(dd * dd) / curv;
        sc0 = (s0 <= t) ? ((1.f - a) * eu + a * hy) : -1e30f;
      }
      {
        float eu = se1[j] * 0.125f;
        float mdot = fmaxf(-sh1[j], 1.000001f);
        float dd = acoshf(mdot);
        float hy = -(dd * dd) / curv;
        sc1 = (s1 <= t) ? ((1.f - a) * eu + a * hy) : -1e30f;
      }
      float rm = fmaxf(hred_max(sc0), hred_max(sc1));
      float nm = fmaxf(m[j], rm);
      float sc = expf(m[j] - nm);
      float p0 = (s0 <= t) ? expf(sc0 - nm) : 0.f;
      float p1 = (s1 <= t) ? expf(sc1 - nm) : 0.f;
      l[j] = l[j] * sc + hred_sum(p0) + hred_sum(p1);
      m[j] = nm;
      acc0[j] *= sc; acc1[j] *= sc; acc2[j] *= sc; acc3[j] *= sc;
      int mr = grp * 8 + j;
      Pt[mr * 32 + ln]      = f2bfu(p0);
      Pt[mr * 32 + 16 + ln] = f2bfu(p1);
    }
    __syncthreads();

    V16U ap;                                           // P as A frag (16x32)
    ap.h[0] = *(const v8bf*)(&Pt[ln * 32 + ka]);
    ap.h[1] = *(const v8bf*)(&Pt[ln * 32 + 8 + ka]);
    const unsigned short* vb = vt + (size_t)bh * D_ * T_ + sb + kc;
    v16bf bv0 = *(const v16bf*)(vb + (size_t)(0 * 16 + ln) * T_);
    v16bf bv1 = *(const v16bf*)(vb + (size_t)(1 * 16 + ln) * T_);
    v16bf bv2 = *(const v16bf*)(vb + (size_t)(2 * 16 + ln) * T_);
    v16bf bv3 = *(const v16bf*)(vb + (size_t)(3 * 16 + ln) * T_);
    acc0 = wmma_bf(ap.v, bv0, acc0);
    acc1 = wmma_bf(ap.v, bv1, acc1);
    acc2 = wmma_bf(ap.v, bv2, acc2);
    acc3 = wmma_bf(ap.v, bv3, acc3);
    __syncthreads();
  }

#pragma unroll
  for (int j = 0; j < 8; j++) {
    int t = qb + grp * 8 + j;
    float sv = spike[(size_t)b * T_ + t];
    float li = sv / fmaxf(l[j], 1e-30f);
    unsigned short* orow = ybf + ((size_t)b * T_ + t) * E_ + hh * D_;
    orow[0 * 16 + ln] = f2bfu(acc0[j] * li);
    orow[1 * 16 + ln] = f2bfu(acc1[j] * li);
    orow[2 * 16 + ln] = f2bfu(acc2[j] * li);
    orow[3 * 16 + ln] = f2bfu(acc3[j] * li);
  }
}

// -------------------------------------------------------- output GEMM
__global__ void __launch_bounds__(256)
k_out_gemm(const unsigned short* __restrict__ ybf,
           const unsigned short* __restrict__ wbf,
           const float* __restrict__ bout,
           float* __restrict__ out) {
  int tile = blockIdx.x * 8 + (threadIdx.x >> 5);   // 2048 tiles
  int lane = threadIdx.x & 31;
  int mtile = tile >> 4;
  int nt64  = tile & 15;
  int rowb = mtile * 16, colb = nt64 * 64;
  int grp = lane >> 4, ln = lane & 15;
  int ka = grp ? 16 : 0, kb16 = grp ? 16 : 0;

  v8f c0 = {}, c1 = {}, c2 = {}, c3 = {};
  const unsigned short* arow = ybf + (size_t)(rowb + ln) * E_;
  for (int kk = 0; kk < E_; kk += 32) {
    __builtin_prefetch(arow + kk + 128, 0, 1);
    V16U a;
    a.h[0] = *(const v8bf*)(arow + kk + ka);
    a.h[1] = *(const v8bf*)(arow + kk + 8 + ka);
    const unsigned short* bb = wbf + (size_t)colb * E_ + kk + kb16;
    v16bf b0 = *(const v16bf*)(bb + (size_t)(0 * 16 + ln) * E_);
    v16bf b1 = *(const v16bf*)(bb + (size_t)(1 * 16 + ln) * E_);
    v16bf b2 = *(const v16bf*)(bb + (size_t)(2 * 16 + ln) * E_);
    v16bf b3 = *(const v16bf*)(bb + (size_t)(3 * 16 + ln) * E_);
    c0 = wmma_bf(a.v, b0, c0);
    c1 = wmma_bf(a.v, b1, c1);
    c2 = wmma_bf(a.v, b2, c2);
    c3 = wmma_bf(a.v, b3, c3);
  }
#pragma unroll
  for (int nt = 0; nt < 4; nt++) {
    v8f c = (nt == 0) ? c0 : (nt == 1) ? c1 : (nt == 2) ? c2 : c3;
#pragma unroll
    for (int j = 0; j < 8; j++) {
      int row = rowb + grp * 8 + j;
      int n = colb + nt * 16 + ln;
      out[(size_t)row * E_ + n] = c[j] + bout[n];
    }
  }
}

// ------------------------------------------------------------- launcher
extern "C" void kernel_launch(void* const* d_in, const int* in_sizes, int n_in,
                              void* d_out, int out_size, void* d_ws, size_t ws_size,
                              hipStream_t stream) {
  (void)in_sizes; (void)n_in; (void)out_size; (void)ws_size;
  const float* x        = (const float*)d_in[0];
  const float* Wqkv     = (const float*)d_in[1];
  const float* bqkv     = (const float*)d_in[2];
  const float* Wout     = (const float*)d_in[3];
  const float* bout     = (const float*)d_in[4];
  const float* Wimp     = (const float*)d_in[5];
  const float* bimp     = (const float*)d_in[6];
  const float* Walpha   = (const float*)d_in[7];
  const float* balpha   = (const float*)d_in[8];
  const float* thr      = (const float*)d_in[9];
  const float* log_k    = (const float*)d_in[10];
  const float* qk_scale = (const float*)d_in[11];

  char* ws = (char*)d_ws;
  size_t off = 0;
  auto alloc = [&](size_t bytes) -> void* {
    void* p = ws + off;
    off += (bytes + 255) & ~(size_t)255;
    return p;
  };
  unsigned short* xbf  = (unsigned short*)alloc((size_t)B_ * T_ * E_ * 2);
  unsigned short* wqbf = (unsigned short*)alloc((size_t)3 * E_ * E_ * 2);
  unsigned short* wobf = (unsigned short*)alloc((size_t)E_ * E_ * 2);
  unsigned short* qbf  = (unsigned short*)alloc((size_t)B_ * H_ * T_ * D_ * 2);
  unsigned short* kbf  = (unsigned short*)alloc((size_t)B_ * H_ * T_ * D_ * 2);
  unsigned short* vt   = (unsigned short*)alloc((size_t)B_ * H_ * D_ * T_ * 2);
  unsigned short* qhbf = (unsigned short*)alloc((size_t)B_ * H_ * T_ * D_ * 2);
  unsigned short* khbf = (unsigned short*)alloc((size_t)B_ * H_ * T_ * D_ * 2);
  unsigned short* ybf  = (unsigned short*)alloc((size_t)B_ * T_ * E_ * 2);
  float* alpha = (float*)alloc((size_t)B_ * T_ * H_ * 4);
  float* spike = (float*)alloc((size_t)B_ * T_ * 4);

  k_f32_to_bf16<<<(B_ * T_ * E_) / 256, 256, 0, stream>>>(x, xbf, B_ * T_ * E_);
  k_f32_to_bf16<<<(3 * E_ * E_) / 256, 256, 0, stream>>>(Wqkv, wqbf, 3 * E_ * E_);
  k_f32_to_bf16<<<(E_ * E_) / 256, 256, 0, stream>>>(Wout, wobf, E_ * E_);

  k_imp_alpha<<<B_ * T_, 256, 0, stream>>>(x, Wimp, bimp, Walpha, balpha, thr,
                                           alpha, spike);

  // 128 mtiles * 48 n64-tiles = 6144 wave-tiles, 8 waves/block
  k_qkv_gemm<<<768, 256, 0, stream>>>(xbf, wqbf, bqkv, qbf, kbf, vt);

  k_hyp<<<(B_ * H_ * T_) / 64, 64, 0, stream>>>(qbf, qhbf, qk_scale, 1);
  k_hyp<<<(B_ * H_ * T_) / 64, 64, 0, stream>>>(kbf, khbf, qk_scale, 0);

  k_attn<<<B_ * H_ * (T_ / 16), 32, 0, stream>>>(qbf, kbf, qhbf, khbf, vt,
                                                 alpha, spike, log_k, ybf);

  // 128 mtiles * 16 n64-tiles = 2048 wave-tiles, 8 waves/block
  k_out_gemm<<<256, 256, 0, stream>>>(ybf, wobf, bout, (float*)d_out);
}
